// cLSTM_4097398801066
// MI455X (gfx1250) — matrix operand order repaired and checked by reading, not compiled, not run on hardware
//
#include <hip/hip_runtime.h>
#include <stdint.h>

// Problem constants (match reference)
#define PNET  32   // networks
#define HID   128  // hidden
#define BATCH 32   // batch
#define TLEN  256  // timesteps
#define NSLAB 5    // K slabs of 32 (K = P + H = 160)

typedef __attribute__((ext_vector_type(16))) __bf16       v16bf;
typedef __attribute__((ext_vector_type(8)))  float        v8f;
typedef __attribute__((ext_vector_type(4)))  unsigned int v4u;
typedef __attribute__((ext_vector_type(4)))  float        v4f;

union Frag {
  v4u          q[2];
  unsigned int u[8];
  v16bf        v;
};

static __device__ __forceinline__ unsigned short f2bf(float x) {
  union { __bf16 b; unsigned short s; } u; u.b = (__bf16)x; return u.s;
}
static __device__ __forceinline__ float bf2f(unsigned short s) {
  union { unsigned int u; float f; } c; c.u = ((unsigned int)s) << 16; return c.f;
}
static __device__ __forceinline__ unsigned int packbf(float lo, float hi) {
  return (unsigned int)f2bf(lo) | ((unsigned int)f2bf(hi) << 16);
}
static __device__ __forceinline__ float sigm(float x) {
  return 1.0f / (1.0f + __expf(-x));
}

// Async DMA: 16 bytes global -> LDS, tracked by ASYNCcnt (no VGPR data traffic).
// GVS mode: saddr = 64-bit SGPR base, vaddr = 32-bit byte offset, vdst = LDS byte addr.
static __device__ __forceinline__ void async_copy16(unsigned int ldsOff,
                                                    const float* base,
                                                    unsigned int gOff) {
  asm volatile("global_load_async_to_lds_b128 %0, %1, %2"
               :: "v"(ldsOff), "v"(gOff), "s"(base)
               : "memory");
}
static __device__ __forceinline__ void wait_asynccnt0() {
  asm volatile("s_wait_asynccnt 0" ::: "memory");
}

// Load one 16-bit WMMA A-fragment (16x32, this lane's slice) from an LDS row.
// p points at (row, khalf-base) in ushorts. 16-bit A layout (ISA 7.12.2):
// VGPR v<4 -> K pair 2v (+8*half), VGPR v>=4 -> K pair 2v+8 (+8*half).
static __device__ __forceinline__ v16bf load_a(const unsigned short* p) {
  Frag f;
#pragma unroll
  for (int v = 0; v < 8; ++v) {
    const int k = (v < 4) ? (2 * v) : (2 * v + 8);
    f.u[v] = *(const unsigned int*)(p + k);
  }
  return f.v;
}

__global__ __launch_bounds__(128, 1)
void clstm_persistent_kernel(const float* __restrict__ X,
                             const float* __restrict__ Wih,
                             const float* __restrict__ Whh,
                             const float* __restrict__ bih,
                             const float* __restrict__ bhh,
                             const float* __restrict__ Wout,
                             const float* __restrict__ bout,
                             float* __restrict__ out)
{
  // --- LDS (~250 KB of the 320 KB/WGP) ---
  __shared__ v4u            sW[160 * 64];        // 160 KB: bf16 [Wih|Whh] pre-swizzled B fragments
  __shared__ float          sG[4][BATCH][132];   // 66 KB: gate pre-activations (i,f,g,o), padded
  __shared__ unsigned short sH[BATCH][136];      // 8.5 KB: h state in bf16, padded rows
  __shared__ unsigned short sX[BATCH][40];       // 2.5 KB: x_t in bf16, padded rows
  __shared__ float          sXraw[BATCH][32];    // 4 KB: fp32 X_{t+1} staging (async DMA target)
  __shared__ float          sPredPart[4][BATCH]; // per-wave head partials (deterministic reduce)
  __shared__ float          sBias[4 * HID];      // b_ih + b_hh
  __shared__ float          sWo[HID];
  __shared__ float          sBo;

  const int n    = blockIdx.x;       // network id
  const int tid  = threadIdx.x;      // 128 threads = 4 wave32
  const int w    = tid >> 5;         // wave id == gate id (i,f,g,o)
  const int lane = tid & 31;
  const int lN   = lane & 15;        // N index inside a 16-wide tile
  const int lH   = lane >> 4;        // K/M half select
  const int half8 = lH * 8;

  // ---------------- init: biases, head weights, zero h ----------------
  for (int k = tid; k < 4 * HID; k += 128) sBias[k] = bih[n * 512 + k] + bhh[n * 512 + k];
  if (tid < HID) sWo[tid] = Wout[n * HID + tid];
  if (tid == 0)  sBo = bout[n];
  for (int k = tid; k < BATCH * 136; k += 128) ((unsigned short*)sH)[k] = 0;

  // ---------------- init: build resident bf16 weight fragments ----------------
  // B matrix (K=160 x N=512): rows 0..31 from W_ih[n][col][k], rows 32..159 from W_hh[n][col][k-32].
  // 16-bit B fragment layout: lane L holds col = base + (L&15), h = L>>4,
  // element i of v16bf = K 8h+i (i<8), 16+8h+(i-8) (i>=8).
  for (int nt = 0; nt < 8; ++nt) {
    const int col = w * 128 + nt * 16 + lN;     // global gate column 0..511
#pragma unroll
    for (int ks = 0; ks < NSLAB; ++ks) {
      const float* src = (ks == 0)
          ? (Wih + ((size_t)n * 512 + col) * 32  + lH * 8)
          : (Whh + ((size_t)n * 512 + col) * 128 + (size_t)(ks - 1) * 32 + lH * 8);
      Frag f;
#pragma unroll
      for (int i = 0; i < 16; ++i) f.v[i] = (__bf16)src[(i < 8) ? i : (i + 8)];
      const int fi = ((w * 8 + nt) * NSLAB + ks) * 64 + lane * 2;
      sW[fi]     = f.q[0];
      sW[fi + 1] = f.q[1];
    }
  }

  // cell state lives entirely in registers: thread owns (b = tid>>2, j = (tid&3)*32 .. +31)
  float creg[32];
#pragma unroll
  for (int i = 0; i < 32; ++i) creg[i] = 0.0f;

  const int eb = tid >> 2;           // elementwise batch row
  const int jb = (tid & 3) * 32;     // elementwise column base
  const int xb = tid >> 2;           // x-stage batch row
  const int xp = (tid & 3) * 8;      // x-stage column base

  // stage X_0 directly into bf16 sX
  {
    const float* xg = X + ((size_t)xb * TLEN + 0) * PNET + xp;
    v4f a = *(const v4f*)xg;
    v4f b = *(const v4f*)(xg + 4);
    unsigned int* dst = (unsigned int*)&sX[xb][xp];
    dst[0] = packbf(a.x, a.y);
    dst[1] = packbf(a.z, a.w);
    dst[2] = packbf(b.x, b.y);
    dst[3] = packbf(b.z, b.w);
  }
  // kick off async DMA of X_1 (fp32) into the staging buffer
  {
    const unsigned goff = (unsigned)((((size_t)xb * TLEN + 1) * PNET + xp) * sizeof(float));
    const unsigned loff = (unsigned)(uintptr_t)&sXraw[xb][xp];
    async_copy16(loff, X, goff);
    async_copy16(loff + 16, X, goff + 16);
  }

  __syncthreads();

  // ---------------- time loop ----------------
  for (int t = 0; t < TLEN; ++t) {
    // ---- gates = [x_t | h] x [Wih | Whh]^T for this wave's gate (128 cols) ----
    v8f acc[2][8];
#pragma unroll
    for (int mt = 0; mt < 2; ++mt)
#pragma unroll
      for (int nt = 0; nt < 8; ++nt)
        acc[mt][nt] = (v8f){0.f, 0.f, 0.f, 0.f, 0.f, 0.f, 0.f, 0.f};

#pragma unroll
    for (int ks = 0; ks < NSLAB; ++ks) {
      const unsigned short* r0;
      const unsigned short* r1;
      if (ks == 0) {
        r0 = &sX[lN][half8];
        r1 = &sX[lN + 16][half8];
      } else {
        const int off = (ks - 1) * 32 + half8;
        r0 = &sH[lN][off];
        r1 = &sH[lN + 16][off];
      }
      const v16bf a0 = load_a(r0);
      const v16bf a1 = load_a(r1);
#pragma unroll
      for (int nt = 0; nt < 8; ++nt) {
        Frag bf;
        const int fi = ((w * 8 + nt) * NSLAB + ks) * 64 + lane * 2;
        bf.q[0] = sW[fi];
        bf.q[1] = sW[fi + 1];
        acc[0][nt] = __builtin_amdgcn_wmma_f32_16x16x32_bf16(
            false, a0, false, bf.v, (short)0, acc[0][nt], false, false);
        acc[1][nt] = __builtin_amdgcn_wmma_f32_16x16x32_bf16(
            false, a1, false, bf.v, (short)0, acc[1][nt], false, false);
      }
    }

    // spill this wave's gate block to LDS (C/D layout: VGPR r -> M = r + 8*(lane>=16))
#pragma unroll
    for (int mt = 0; mt < 2; ++mt)
#pragma unroll
      for (int nt = 0; nt < 8; ++nt)
#pragma unroll
        for (int r = 0; r < 8; ++r)
          sG[w][mt * 16 + lH * 8 + r][nt * 16 + lN] = acc[mt][nt][r];

    // X_{t+1} DMA (issued last step) must have landed before the repack below
    wait_asynccnt0();
    __syncthreads();   // all four gates available; sXraw ready

    // ---- elementwise LSTM cell update; c in registers, h -> bf16 LDS ----
#pragma unroll
    for (int p2 = 0; p2 < 16; ++p2) {
      const int j0 = jb + 2 * p2;
      const int j1 = j0 + 1;
      const float i0 = sigm(sG[0][eb][j0] + sBias[j0]);
      const float f0 = sigm(sG[1][eb][j0] + sBias[HID + j0]);
      const float g0 = tanhf(sG[2][eb][j0] + sBias[2 * HID + j0]);
      const float o0 = sigm(sG[3][eb][j0] + sBias[3 * HID + j0]);
      const float i1 = sigm(sG[0][eb][j1] + sBias[j1]);
      const float f1 = sigm(sG[1][eb][j1] + sBias[HID + j1]);
      const float g1 = tanhf(sG[2][eb][j1] + sBias[2 * HID + j1]);
      const float o1 = sigm(sG[3][eb][j1] + sBias[3 * HID + j1]);
      const float c0 = f0 * creg[2 * p2]     + i0 * g0;
      const float c1 = f1 * creg[2 * p2 + 1] + i1 * g1;
      creg[2 * p2]     = c0;
      creg[2 * p2 + 1] = c1;
      const float h0 = o0 * tanhf(c0);
      const float h1 = o1 * tanhf(c1);
      *(unsigned int*)&sH[eb][j0] = packbf(h0, h1);
    }

    // repack async-staged fp32 X_{t+1} into bf16 sX for the next step
    if (t + 1 < TLEN) {
      unsigned int* dst = (unsigned int*)&sX[xb][xp];
      dst[0] = packbf(sXraw[xb][xp + 0], sXraw[xb][xp + 1]);
      dst[1] = packbf(sXraw[xb][xp + 2], sXraw[xb][xp + 3]);
      dst[2] = packbf(sXraw[xb][xp + 4], sXraw[xb][xp + 5]);
      dst[3] = packbf(sXraw[xb][xp + 6], sXraw[xb][xp + 7]);
    }
    __syncthreads();   // h + next-step sX ready; sXraw reads done

    // issue async DMA of X_{t+2} into the (now free) staging buffer
    if (t + 2 < TLEN) {
      const unsigned goff = (unsigned)((((size_t)xb * TLEN + (t + 2)) * PNET + xp) * sizeof(float));
      const unsigned loff = (unsigned)(uintptr_t)&sXraw[xb][xp];
      async_copy16(loff, X, goff);
      async_copy16(loff + 16, X, goff + 16);
    }

    // ---- 1x1 conv head: 4-way split partial dots (deterministic, no atomics) ----
    {
      float s = 0.0f;
      const unsigned short* hr = &sH[lane][w * 32];
#pragma unroll
      for (int j = 0; j < 32; ++j) s += bf2f(hr[j]) * sWo[w * 32 + j];
      sPredPart[w][lane] = s;
    }
    __syncthreads();

    if (w == 0) {
      const float s = sBo + sPredPart[0][lane] + sPredPart[1][lane]
                          + sPredPart[2][lane] + sPredPart[3][lane];
      out[((size_t)lane * TLEN + t) * PNET + n] = s;
    }
  }
}

extern "C" void kernel_launch(void* const* d_in, const int* in_sizes, int n_in,
                              void* d_out, int out_size, void* d_ws, size_t ws_size,
                              hipStream_t stream) {
  (void)in_sizes; (void)n_in; (void)out_size; (void)d_ws; (void)ws_size;
  const float* X    = (const float*)d_in[0];
  const float* Wih  = (const float*)d_in[1];
  const float* Whh  = (const float*)d_in[2];
  const float* bih  = (const float*)d_in[3];
  const float* bhh  = (const float*)d_in[4];
  const float* Wout = (const float*)d_in[5];
  const float* bout = (const float*)d_in[6];
  clstm_persistent_kernel<<<dim3(PNET), dim3(128), 0, stream>>>(
      X, Wih, Whh, bih, bhh, Wout, bout, (float*)d_out);
}
